// CombinedEmbedding_13005160972681
// MI455X (gfx1250) — matrix operand order repaired and checked by reading, compiled, not run on hardware
//
#include <hip/hip_runtime.h>
#include <hip/hip_bf16.h>
#include <math.h>

// ---------------------------------------------------------------------------
// CombinedEmbedding fused implementation for gfx1250 (MI455X), wave32 + WMMA.
//
//   out = e_z + resmlp(av_charge) + resmlp(av_spin)
//
// Restructuring:
//  * dot(q,k±) = e_z·(lin_w k±) + lin_b·k±  -> q GEMM eliminated (prep kernel).
//  * w1/w2/w3 pre-packed ONCE to bf16 in the exact WMMA B-fragment layout
//    (panel-major [kp][n][k]) -> mol_kernel loads B fragments straight from
//    L2/WGP$-resident global memory: no per-block conversion, no panel LDS
//    staging, no per-panel barriers.
//  * charge & spin embeddings stacked into one 64x256 A matrix.
//  * av is rank-1 (a_i ⊗ v) -> recomputed in the GEMM2 epilogue.
//  * sigmoid via v_rcp_f32 (no IEEE division fixup chains).
// ---------------------------------------------------------------------------

typedef __attribute__((ext_vector_type(16))) __bf16 v16bf;
typedef __attribute__((ext_vector_type(8)))  __bf16 v8bf;
typedef __attribute__((ext_vector_type(8)))  float  v8f;

union Frag { v16bf v; v8bf h[2]; };

#define FEAT 256
#define N_MOL 2048
#define ROWS 32             // atoms per molecule
#define AROWS 64            // stacked A matrix rows (charge ++ spin)
#define SW 264              // bf16 stride of activation buffer (row = 528B = 33*16)
#define PANEL_ELEMS 8192    // one packed K-panel: 256 cols x 32 k (bf16)
#define WPACK_ELEMS 65536   // one packed weight: 8 panels
#define WS_PACK_BYTE_OFF 4096

__device__ __forceinline__ float sigm_(float x) {
    return __builtin_amdgcn_rcpf(1.f + __expf(-x));     // v_rcp_f32, no div fixup
}
__device__ __forceinline__ float softplusf_(float x) {
    return x > 20.f ? x : log1pf(__expf(x));
}

// One [64 x 256] @ [256 x 256] GEMM, K tiled in panels of 32.
// A (bf16) in LDS (stride SW). B fragments come directly from the packed
// bf16 weight (global, panel-major), two global_load_b128 per 16x16 tile.
// Wave layout: rb = wave&3 -> rows rb*16.. ; cg = wave>>2 -> cols cg*128..
__device__ __forceinline__ void gemm_direct(const __bf16* __restrict__ WT,
                                            const __bf16* sA,
                                            int tid, int rb, int cg, v8f acc[8])
{
    const int lane   = tid & 31;
    const int laneLo = lane & 15;
    const int hi     = lane >> 4;   // 0: lanes 0-15, 1: lanes 16-31

    for (int kp = 0; kp < 8; ++kp) {
        // A fragment per ISA 16-bit A layout:
        //  lanes 0-15 : row M=lane,    K = kk+0..7  and kk+16..23
        //  lanes16-31 : row M=lane-16, K = kk+8..15 and kk+24..31
        Frag a;
        const __bf16* ap = sA + (rb * 16 + laneLo) * SW + kp * 32 + hi * 8;
        a.h[0] = *(const v8bf*)(ap);
        a.h[1] = *(const v8bf*)(ap + 16);

        // Packed B: element (kp, n, k_local) at WT[kp*8192 + n*32 + k_local].
        // Fragment: lanes 0-15 col N=lane hold K=kk+0..7 and kk+16..23,
        //           lanes16-31 col N=lane-16 hold K=kk+8..15 and kk+24..31.
        const __bf16* pb = WT + kp * PANEL_ELEMS + (cg * 128 + laneLo) * 32 + hi * 8;
        if (kp < 7)  // prefetch next packed panel (64B per thread covers 16KB)
            __builtin_prefetch(WT + (kp + 1) * PANEL_ELEMS + tid * 32, 0, 3);

        #pragma unroll
        for (int t = 0; t < 8; ++t) {
            const __bf16* bp = pb + t * 512;     // +16 columns
            Frag b;
            b.h[0] = *(const v8bf*)(bp);
            b.h[1] = *(const v8bf*)(bp + 16);
            acc[t] = __builtin_amdgcn_wmma_f32_16x16x32_bf16(
                false, a.v, false, b.v, (short)0, acc[t], false, false);
        }
    }
}

// ---------------------------------------------------------------------------
// Kernel 0a: u± = lin_w @ k±, c± = lin_b·k±  (tiny)
// ws floats: [0,256) u_plus | [256,512) u_minus | 512 cP | 513 cM
// ---------------------------------------------------------------------------
__global__ __launch_bounds__(256)
void prep_kernel(const float* __restrict__ lin_w, const float* __restrict__ lin_b,
                 const float* __restrict__ k_plus, const float* __restrict__ k_minus,
                 float* __restrict__ ws)
{
    __shared__ float sp[256], sm[256];
    int i = threadIdx.x;
    float up = 0.f, um = 0.f;
    for (int c = 0; c < FEAT; ++c) {
        float w = lin_w[i * FEAT + c];
        up = fmaf(w, k_plus[c], up);
        um = fmaf(w, k_minus[c], um);
    }
    ws[i]       = up;
    ws[256 + i] = um;
    sp[i] = lin_b[i] * k_plus[i];
    sm[i] = lin_b[i] * k_minus[i];
    __syncthreads();
    for (int s = 128; s > 0; s >>= 1) {
        if (i < s) { sp[i] += sp[i + s]; sm[i] += sm[i + s]; }
        __syncthreads();
    }
    if (i == 0) { ws[512] = sp[0]; ws[513] = sm[0]; }
}

// ---------------------------------------------------------------------------
// Kernel 0b: pack w1|w2|w3 (f32, [K][N]) -> bf16 panel-major fragment layout:
//   pack[w][kp][n][kk] = W_w[(kp*32+kk)*256 + n]
// One-time 768KB -> 384KB; contiguous coalesced writes.
// ---------------------------------------------------------------------------
__global__ __launch_bounds__(256)
void pack_kernel(const float* __restrict__ w1, const float* __restrict__ w2,
                 const float* __restrict__ w3, __bf16* __restrict__ pack)
{
    int idx = blockIdx.x * 256 + threadIdx.x;      // 3*65536 total
    int w   = idx >> 16;
    int r   = idx & 0xFFFF;
    int kp  = r >> 13;
    int q   = r & 0x1FFF;
    int n   = q >> 5;
    int kk  = q & 31;
    const float* W = (w == 0) ? w1 : (w == 1) ? w2 : w3;
    pack[idx] = (__bf16)W[(kp * 32 + kk) * FEAT + n];
}

// ---------------------------------------------------------------------------
// Kernel 1: e_z = elec_config[z] @ m_mat_w + z_embed[z]  ->  out (f32)
// ---------------------------------------------------------------------------
__global__ __launch_bounds__(256)
void ez_kernel(const int* __restrict__ z, const float* __restrict__ ec,
               const float* __restrict__ m, const float* __restrict__ zemb,
               float* __restrict__ out)
{
    const int c = threadIdx.x;
    const int rowBase = blockIdx.x * ROWS;
    for (int r = 0; r < ROWS; ++r) {
        int zi = z[rowBase + r];
        float acc = zemb[zi * FEAT + c];
        const float* e = ec + zi * 20;      // block-uniform -> scalar loads
        #pragma unroll
        for (int j = 0; j < 20; ++j)
            acc = fmaf(e[j], m[j * FEAT + c], acc);
        out[(size_t)(rowBase + r) * FEAT + c] = acc;
    }
}

// ---------------------------------------------------------------------------
// Kernel 2: fused attention scalars + both residual MLPs, one molecule/block.
// A rows 0..31 = charge embedding, rows 32..63 = spin embedding.
// ---------------------------------------------------------------------------
__global__ __launch_bounds__(256)
void mol_kernel(const float* __restrict__ charge, const float* __restrict__ spin,
                const float* __restrict__ ws, const __bf16* __restrict__ pack,
                const float* __restrict__ v_plus, const float* __restrict__ v_minus,
                const float* __restrict__ a1, const float* __restrict__ b1,
                const float* __restrict__ a2, const float* __restrict__ b2,
                const float* __restrict__ a3, const float* __restrict__ b3,
                float* __restrict__ out)
{
    __shared__ __attribute__((aligned(16))) __bf16 sWork[AROWS * SW];  // A operand
    __shared__ float sDotP[ROWS], sDotM[ROWS], sA64[AROWS];

    const int tid = threadIdx.x;
    const int mol = blockIdx.x;
    const int rowBase = mol * ROWS;
    const int lane = tid & 31;
    const int wave = tid >> 5;
    const int rb = wave & 3;    // row block: rows rb*16 .. rb*16+15 (rb>=2 => spin)
    const int cg = wave >> 2;   // col group: cols cg*128 .. cg*128+127
    const int laneLo = lane & 15, hi = lane >> 4;

    const float chg = charge[mol], spn = spin[mol];
    const float* vq = (chg >= 0.f) ? v_plus : v_minus;
    const float* vs = (spn >= 0.f) ? v_plus : v_minus;

    // ---- P1: per-row dots e_z . u± (8 lanes per row), then a_i per psi ----
    {
        const float* up = ws;
        const float* um = ws + 256;
        int r  = tid >> 3;
        int c0 = (tid & 7) * 32;
        const float* ez = out + (size_t)(rowBase + r) * FEAT;  // e_z from kernel 1
        float dp = 0.f, dm = 0.f;
        #pragma unroll
        for (int c = 0; c < 32; ++c) {
            float e = ez[c0 + c];
            dp = fmaf(e, up[c0 + c], dp);
            dm = fmaf(e, um[c0 + c], dm);
        }
        #pragma unroll
        for (int off = 4; off > 0; off >>= 1) {
            dp += __shfl_xor(dp, off, 32);
            dm += __shfl_xor(dm, off, 32);
        }
        if ((lane & 7) == 0) { sDotP[r] = dp; sDotM[r] = dm; }
    }
    __syncthreads();
    if (tid < 32) {
        float cP = ws[512], cM = ws[513];
        float argq = ((chg >= 0.f) ? (sDotP[tid] + cP) : (sDotM[tid] + cM)) * 0.0625f;
        float args = ((spn >= 0.f) ? (sDotP[tid] + cP) : (sDotM[tid] + cM)) * 0.0625f;
        float nq = softplusf_(argq);
        float ns = softplusf_(args);
        float dq = nq, dsn = ns;        // segment sums over the 32 atoms
        #pragma unroll
        for (int off = 16; off > 0; off >>= 1) {
            dq  += __shfl_xor(dq,  off, 32);
            dsn += __shfl_xor(dsn, off, 32);
        }
        sA64[tid]        = chg * nq / dq;
        sA64[tid + ROWS] = spn * ns / dsn;
    }
    __syncthreads();

    // ---- build A = swish1(av) for both embeddings (64 rows, c = tid) ----
    {
        const float vqc = vq[tid], vsc = vs[tid];
        const float A1 = a1[tid], B1 = b1[tid];
        #pragma unroll 4
        for (int r = 0; r < AROWS; ++r) {
            float av = sA64[r] * (r < ROWS ? vqc : vsc);
            sWork[r * SW + tid] = (__bf16)(A1 * av * sigm_(B1 * av));
        }
    }
    __syncthreads();

    v8f acc[8];

    // ---- GEMM1: swish1(av) @ w1 ----
    #pragma unroll
    for (int t = 0; t < 8; ++t) acc[t] = (v8f){0,0,0,0,0,0,0,0};
    gemm_direct(pack, sWork, tid, rb, cg, acc);
    __syncthreads();
    #pragma unroll
    for (int t = 0; t < 8; ++t) {                 // epi1: swish2 -> sWork
        int col = (cg * 8 + t) * 16 + laneLo;
        float A2 = a2[col], B2 = b2[col];
        #pragma unroll
        for (int r2 = 0; r2 < 8; ++r2) {
            int row = rb * 16 + r2 + hi * 8;      // C layout: lanes16-31 are M+8
            float x = acc[t][r2];
            sWork[row * SW + col] = (__bf16)(A2 * x * sigm_(B2 * x));
        }
    }
    __syncthreads();

    // ---- GEMM2: swish2(.) @ w2 ----
    #pragma unroll
    for (int t = 0; t < 8; ++t) acc[t] = (v8f){0,0,0,0,0,0,0,0};
    gemm_direct(pack + WPACK_ELEMS, sWork, tid, rb, cg, acc);
    __syncthreads();
    {   // epi2: h = av + x (av recomputed, rank-1); swish3 -> sWork
        const float* vv = (rb >= 2) ? vs : vq;    // embedding uniform per wave
        #pragma unroll
        for (int t = 0; t < 8; ++t) {
            int col = (cg * 8 + t) * 16 + laneLo;
            float A3 = a3[col], B3 = b3[col], vc = vv[col];
            #pragma unroll
            for (int r2 = 0; r2 < 8; ++r2) {
                int row = rb * 16 + r2 + hi * 8;
                float h = sA64[row] * vc + acc[t][r2];
                sWork[row * SW + col] = (__bf16)(A3 * h * sigm_(B3 * h));
            }
        }
    }
    __syncthreads();

    // ---- GEMM3: swish3(h) @ w3 ----
    #pragma unroll
    for (int t = 0; t < 8; ++t) acc[t] = (v8f){0,0,0,0,0,0,0,0};
    gemm_direct(pack + 2 * WPACK_ELEMS, sWork, tid, rb, cg, acc);
    __syncthreads();

    // ---- epi3: combine charge+spin contributions, out += ----
    // Spin waves (rb>=2) park tiles in LDS (f32, reusing sWork storage);
    // charge waves fold both into the single global RMW. No atomics needed.
    float* sAccF = (float*)sWork;                 // 32*SW floats == sWork bytes
    if (rb >= 2) {
        #pragma unroll
        for (int t = 0; t < 8; ++t) {
            int col = (cg * 8 + t) * 16 + laneLo;
            #pragma unroll
            for (int r2 = 0; r2 < 8; ++r2) {
                int row = rb * 16 + r2 + hi * 8;  // 32..63
                sAccF[(row - ROWS) * SW + col] = acc[t][r2];
            }
        }
    }
    __syncthreads();
    if (rb < 2) {
        #pragma unroll
        for (int t = 0; t < 8; ++t) {
            int col = (cg * 8 + t) * 16 + laneLo;
            #pragma unroll
            for (int r2 = 0; r2 < 8; ++r2) {
                int row = rb * 16 + r2 + hi * 8;  // 0..31
                size_t idx = (size_t)(rowBase + row) * FEAT + col;
                out[idx] += acc[t][r2] + sAccF[row * SW + col];
            }
        }
    }
}

// ---------------------------------------------------------------------------
extern "C" void kernel_launch(void* const* d_in, const int* in_sizes, int n_in,
                              void* d_out, int out_size, void* d_ws, size_t ws_size,
                              hipStream_t stream)
{
    const float* charge  = (const float*)d_in[0];
    const float* spin    = (const float*)d_in[1];
    const int*   z       = (const int*)  d_in[2];
    // d_in[3] num_atoms: constant 32, folded into indexing
    const float* ec      = (const float*)d_in[4];
    const float* mmat    = (const float*)d_in[5];
    const float* zemb    = (const float*)d_in[6];
    const float* lin_w   = (const float*)d_in[7];
    const float* lin_b   = (const float*)d_in[8];
    const float* k_plus  = (const float*)d_in[9];
    const float* k_minus = (const float*)d_in[10];
    const float* v_plus  = (const float*)d_in[11];
    const float* v_minus = (const float*)d_in[12];
    const float* w1      = (const float*)d_in[13];
    const float* w2      = (const float*)d_in[14];
    const float* w3      = (const float*)d_in[15];
    const float* a1      = (const float*)d_in[16];
    const float* b1      = (const float*)d_in[17];
    const float* a2      = (const float*)d_in[18];
    const float* b2      = (const float*)d_in[19];
    const float* a3      = (const float*)d_in[20];
    const float* b3      = (const float*)d_in[21];
    float* out = (float*)d_out;
    float* wsf = (float*)d_ws;
    __bf16* pack = (__bf16*)((char*)d_ws + WS_PACK_BYTE_OFF);   // 384KB packed weights

    prep_kernel<<<1, 256, 0, stream>>>(lin_w, lin_b, k_plus, k_minus, wsf);
    pack_kernel<<<3 * 256, 256, 0, stream>>>(w1, w2, w3, pack);
    ez_kernel<<<N_MOL, 256, 0, stream>>>(z, ec, mmat, zemb, out);
    mol_kernel<<<N_MOL, 256, 0, stream>>>(charge, spin, wsf, pack, v_plus, v_minus,
                                          a1, b1, a2, b2, a3, b3, out);
}
